// SingleBNN_1331439862018
// MI455X (gfx1250) — 8-bit matrix operand order repaired and checked by reading, compiled, not run on hardware
//
#include <hip/hip_runtime.h>
#include <hip/hip_bf16.h>
#include <cstdint>

typedef __attribute__((ext_vector_type(8))) int v8i;

__global__ __launch_bounds__(256)
void conv0_kernel(const float* __restrict__ x, const float* __restrict__ w,
                  float* __restrict__ accf, int total)
{
    int idx = blockIdx.x * 256 + threadIdx.x;
    if (idx >= total) return;
    int n  = idx >> 12;
    int p  = idx & 4095;
    int py = p >> 6, px = p & 63;
    const float* xn = x + (size_t)n * 3 * 4096;
    float xv[27];
#pragma unroll
    for (int ci = 0; ci < 3; ++ci)
#pragma unroll
        for (int t = 0; t < 9; ++t) {
            int yy = py + t / 3 - 1, xx = px + t % 3 - 1;
            float v = 0.f;
            if (yy >= 0 && yy < 64 && xx >= 0 && xx < 64)
                v = xn[(ci * 64 + yy) * 64 + xx];
            xv[ci * 9 + t] = v;
        }
    float* o = accf + (size_t)idx * 16;
#pragma unroll
    for (int c = 0; c < 16; ++c) {
        float s = 0.f;
#pragma unroll
        for (int k = 0; k < 27; ++k) s += xv[k] * w[c * 27 + k];
        o[c] = s;
    }
}

__global__ __launch_bounds__(256)
void prep_kernel(const float* __restrict__ w, int8_t* __restrict__ wp,
                 int Cin, int Cout, int nkc, int total)
{
    int idx = blockIdx.x * 256 + threadIdx.x;
    if (idx >= total) return;
    int b    = idx & 3;
    int v    = (idx >> 2) & 7;
    int lane = (idx >> 5) & 31;
    int rest = idx >> 10;
    int kc   = rest % nkc;  rest /= nkc;
    int t    = rest % 9;
    int tile = rest / 9;
    int K  = ((v >> 1) << 4) + ((lane >> 4) << 3) + ((v & 1) << 2) + b;
    int o  = tile * 16 + (lane & 15);
    int ci = kc * 64 + K;
    int8_t s = 0;
    if (o < Cout && ci < Cin) {
        float wv = w[(o * Cin + ci) * 9 + t];
        s = (int8_t)((wv > 0.f) - (wv < 0.f));
    }
    wp[idx] = s;
}

template <int NT>
__global__ __launch_bounds__(128)
void bconv_kernel(const int8_t* __restrict__ act, const int8_t* __restrict__ wp,
                  int* __restrict__ acc,
                  int H, int W, int cstride, int nkc, int accC)
{
    const int lane  = threadIdx.x & 31;
    const int wave  = threadIdx.x >> 5;
    const int n     = blockIdx.z;
    const int P     = H * W;
    const int pbase = (blockIdx.x * 4 + wave) * 16;
    const int m     = lane & 15;
    const int p     = pbase + m;
    const int py    = p / W, px = p % W;
    const int sel8  = (lane >> 4) << 3;
    const int t0    = blockIdx.y * NT;
    const size_t nbase = (size_t)n * P;
    const size_t wtile = (size_t)9 * nkc * 1024;
    const int8_t* wbase = wp + (size_t)t0 * wtile + (size_t)lane * 32;

    v8i c4[NT] = {};

#pragma unroll
    for (int t = 0; t < 9; ++t) {
        const int dy = t / 3 - 1;
        const int dx = t % 3 - 1;
        const int yy = py + dy;
        const int xx = px + dx;
        const bool ok = (yy >= 0) && (yy < H) && (xx >= 0) && (xx < W);
        const int8_t* ap = ok
            ? (act + (nbase + (size_t)(yy * W + xx)) * cstride + sel8)
            : nullptr;
        for (int kc = 0; kc < nkc; ++kc) {
            union { unsigned long long q[4]; v8i v; } A;
            if (ok) {
                const unsigned long long* aq =
                    (const unsigned long long*)(ap + (kc << 6));
                A.q[0] = aq[0];
                A.q[1] = aq[2];
                A.q[2] = aq[4];
                A.q[3] = aq[6];
            } else {
                A.q[0] = 0; A.q[1] = 0; A.q[2] = 0; A.q[3] = 0;
            }
            const int8_t* bp = wbase + (size_t)(t * nkc + kc) * 1024;
            v8i bf[NT];
#pragma unroll
            for (int j = 0; j < NT; ++j)
                bf[j] = *(const v8i*)(bp + (size_t)j * wtile);
#pragma unroll
            for (int j = 0; j < NT; ++j)
                c4[j] = __builtin_amdgcn_wmma_i32_16x16x64_iu8(
                    true, A.v, true, bf[j], c4[j], false, false);
        }
    }

    const int cl   = lane & 15;
    const int mrow = (lane >> 4) << 3;
#pragma unroll
    for (int j = 0; j < NT; ++j) {
        const int cch = (t0 + j) * 16 + cl;
#pragma unroll
        for (int r = 0; r < 8; ++r) {
            const int pix = pbase + mrow + r;
            acc[(nbase + pix) * accC + cch] = c4[j][r];
        }
    }
}

template <typename T>
__global__ __launch_bounds__(256)
void stats_kernel(const T* __restrict__ acc, float* __restrict__ stats,
                  int npix, int accC)
{
    int tid  = blockIdx.x * 256 + threadIdx.x;
    int tcnt = gridDim.x * 256;
    int c    = tid % accC;
    int rows = tcnt / accC;
    int r0   = tid / accC;
    float s = 0.f, ss = 0.f;
    for (int pix = r0; pix < npix; pix += rows) {
        float v = (float)acc[(size_t)pix * accC + c];
        s += v; ss += v * v;
    }
    atomicAdd(&stats[c], s);
    atomicAdd(&stats[accC + c], ss);
}

__global__ void bnparams_kernel(const float* __restrict__ stats,
                                const float* __restrict__ g,
                                const float* __restrict__ be,
                                float* __restrict__ sAB,
                                float cntInv, int accC, int Cout)
{
    int c = threadIdx.x;
    if (c >= accC) return;
    float mean = stats[c] * cntInv;
    float var  = stats[accC + c] * cntInv - mean * mean;
    float rs   = rsqrtf(var + 1e-5f);
    float sA = 0.f, sB = 0.f;
    if (c < Cout) { sA = g[c] * rs; sB = be[c] - sA * mean; }
    sAB[c] = sA; sAB[accC + c] = sB;
}

template <typename T, bool POOL>
__global__ __launch_bounds__(256)
void binarize_kernel(const T* __restrict__ acc, const float* __restrict__ sAB,
                     int8_t* __restrict__ out, int H, int W,
                     int accC, int Cout, int ostride, size_t total)
{
    size_t idx = (size_t)blockIdx.x * 256 + threadIdx.x;
    if (idx >= total) return;
    int cs   = (int)(idx % ostride);
    size_t q = idx / ostride;
    int8_t r = 0;
    if (cs < Cout) {
        float sA = sAB[cs], sB = sAB[accC + cs];
        if (!POOL) {
            float v = sA * (float)acc[q * accC + cs] + sB;
            r = (int8_t)((v > 0.f) - (v < 0.f));
        } else {
            int Wo = W >> 1, Ho = H >> 1;
            int xo = (int)(q % Wo); size_t q2 = q / Wo;
            int yo = (int)(q2 % Ho); int n = (int)(q2 / Ho);
            int8_t mx = -2;
#pragma unroll
            for (int d = 0; d < 4; ++d) {
                int yy = 2 * yo + (d >> 1), xx = 2 * xo + (d & 1);
                size_t pix = ((size_t)n * H + yy) * W + xx;
                float v = sA * (float)acc[pix * accC + cs] + sB;
                int8_t s = (int8_t)((v > 0.f) - (v < 0.f));
                mx = (s > mx) ? s : mx;
            }
            r = mx;
        }
    }
    out[idx] = r;
}

__global__ __launch_bounds__(32)
void final_kernel(const int* __restrict__ acc, const float* __restrict__ sAB,
                  float* __restrict__ out)
{
    int n = blockIdx.x, lane = threadIdx.x;
    float z[10];
#pragma unroll
    for (int c = 0; c < 10; ++c) z[c] = 0.f;
    for (int p = lane; p < 64; p += 32) {
        const int* a = acc + ((size_t)n * 64 + p) * 16;
#pragma unroll
        for (int c = 0; c < 10; ++c)
            z[c] += sAB[c] * (float)a[c] + sAB[16 + c];
    }
#pragma unroll
    for (int c = 0; c < 10; ++c)
#pragma unroll
        for (int off = 16; off >= 1; off >>= 1)
            z[c] += __shfl_xor(z[c], off, 32);
    if (lane == 0) {
        float m = -1e30f;
#pragma unroll
        for (int c = 0; c < 10; ++c) { z[c] *= (1.f / 64.f); m = fmaxf(m, z[c]); }
        float s = 0.f;
#pragma unroll
        for (int c = 0; c < 10; ++c) s += __expf(z[c] - m);
        float lse = m + __logf(s);
#pragma unroll
        for (int c = 0; c < 10; ++c) out[n * 10 + c] = z[c] - lse;
    }
}

extern "C" void kernel_launch(void* const* d_in, const int* in_sizes, int n_in,
                              void* d_out, int out_size, void* d_ws, size_t ws_size,
                              hipStream_t stream)
{
    (void)in_sizes; (void)n_in; (void)out_size; (void)ws_size;

    static const int CIN[10]  = {3, 16, 16, 32, 32, 64, 64, 128, 128, 256};
    static const int COUT[10] = {16, 16, 32, 32, 64, 64, 128, 128, 256, 10};
    const int N = 256;

    const float* x = (const float*)d_in[0];
    auto Wp  = [&](int i) { return (const float*)d_in[1 + 4 * i]; };
    auto Gp  = [&](int i) { return (const float*)d_in[3 + 4 * i]; };
    auto BEp = [&](int i) { return (const float*)d_in[4 + 4 * i]; };

    char*  ws    = (char*)d_ws;
    float* stats = (float*)ws;
    float* sAB   = (float*)(ws + 2048);
    size_t woff[10];
    size_t wcur = 4096;
    for (int i = 1; i <= 9; ++i) {
        int accC   = ((COUT[i] + 15) / 16) * 16;
        int ntiles = accC / 16;
        int nkc    = (CIN[i] + 63) / 64;
        woff[i] = wcur;
        wcur += (size_t)ntiles * 9 * nkc * 1024;
    }
    size_t accOff  = (wcur + 255) & ~(size_t)255;
    size_t accBy   = (size_t)N * 4096 * 32 * 4;
    size_t actBy   = (size_t)N * 4096 * 64;
    int*    acc  = (int*)(ws + accOff);
    float*  accf = (float*)(ws + accOff);
    int8_t* actA = (int8_t*)(ws + accOff + accBy);
    int8_t* actB = (int8_t*)(ws + accOff + accBy + actBy);

    hipMemsetAsync(stats, 0, 2 * 16 * sizeof(float), stream);
    conv0_kernel<<<N * 4096 / 256, 256, 0, stream>>>(x, Wp(0), accf, N * 4096);
    stats_kernel<float><<<512, 256, 0, stream>>>(accf, stats, N * 4096, 16);
    bnparams_kernel<<<1, 16, 0, stream>>>(stats, Gp(0), BEp(0), sAB,
                                          1.f / (N * 4096.f), 16, 16);
    {
        size_t tot = (size_t)N * 4096 * 64;
        binarize_kernel<float, false><<<(unsigned)((tot + 255) / 256), 256, 0, stream>>>(
            accf, sAB, actA, 64, 64, 16, 16, 64, tot);
    }

    int8_t* cur = actA; int8_t* nxt = actB;
    int H = 64;
    for (int i = 1; i <= 9; ++i) {
        int Cin = CIN[i], Cout = COUT[i];
        int accC    = ((Cout + 15) / 16) * 16;
        int ntiles  = accC / 16;
        int nkc     = (Cin + 63) / 64;
        int cstride = Cin < 64 ? 64 : Cin;
        int W = H, P = H * W;
        int8_t* wpk = (int8_t*)(ws + woff[i]);

        int ptot = ntiles * 9 * nkc * 1024;
        prep_kernel<<<(ptot + 255) / 256, 256, 0, stream>>>(Wp(i), wpk, Cin, Cout,
                                                            nkc, ptot);
        hipMemsetAsync(stats, 0, 2 * accC * sizeof(float), stream);

        int NT = (ntiles >= 4) ? 4 : ntiles;
        dim3 grid(P / 64, ntiles / NT, N);
        if (NT == 4)
            bconv_kernel<4><<<grid, 128, 0, stream>>>(cur, wpk, acc, H, W,
                                                      cstride, nkc, accC);
        else if (NT == 2)
            bconv_kernel<2><<<grid, 128, 0, stream>>>(cur, wpk, acc, H, W,
                                                      cstride, nkc, accC);
        else
            bconv_kernel<1><<<grid, 128, 0, stream>>>(cur, wpk, acc, H, W,
                                                      cstride, nkc, accC);

        stats_kernel<int><<<512, 256, 0, stream>>>(acc, stats, N * P, accC);
        bnparams_kernel<<<1, accC, 0, stream>>>(stats, Gp(i), BEp(i), sAB,
                                                1.f / ((float)N * P), accC, Cout);
        if (i < 9) {
            int ostride = Cout < 64 ? 64 : Cout;
            bool pool = (i == 3 || i == 5 || i == 7);
            if (pool) {
                size_t tot = (size_t)N * (H / 2) * (W / 2) * ostride;
                binarize_kernel<int, true><<<(unsigned)((tot + 255) / 256), 256, 0, stream>>>(
                    acc, sAB, nxt, H, W, accC, Cout, ostride, tot);
                H /= 2;
            } else {
                size_t tot = (size_t)N * P * ostride;
                binarize_kernel<int, false><<<(unsigned)((tot + 255) / 256), 256, 0, stream>>>(
                    acc, sAB, nxt, H, W, accC, Cout, ostride, tot);
            }
            int8_t* t = cur; cur = nxt; nxt = t;
        } else {
            final_kernel<<<N, 32, 0, stream>>>(acc, sAB, (float*)d_out);
        }
    }
}
